// LayerNormDropoutLSTMCell_88587995447884
// MI455X (gfx1250) — matrix operand order repaired and checked by reading, not compiled, not run on hardware
//
#include <hip/hip_runtime.h>
#include <stdint.h>

// ---------------------------------------------------------------------------
// Types for CDNA5 WMMA (wave32): v16bf operands, v8f accumulator
// ---------------------------------------------------------------------------
typedef __attribute__((ext_vector_type(16))) __bf16 v16bf;
typedef __attribute__((ext_vector_type(8)))  float  v8f;

union FragAB {
    v16bf v;
    uint4 q[2];   // 32 bytes total, loaded as 2x b128 from LDS
};

// f32 -> bf16 round-to-nearest-even
__device__ __forceinline__ unsigned f2bf(float f) {
    unsigned u = __float_as_uint(f);
    unsigned r = u + 0x7FFFu + ((u >> 16) & 1u);
    return r >> 16;
}

// CDNA5 async copy: global -> LDS, 16B per lane, no VGPR data path.
// GVS addressing: mem = SADDR(64) + VADDR(32) ; LDS dest offset in VGPR.
__device__ __forceinline__ void async_ld16(unsigned lds_off, unsigned goff,
                                           const void* base) {
    asm volatile("global_load_async_to_lds_b128 %0, %1, %2"
                 :: "v"(lds_off), "v"(goff), "s"(base)
                 : "memory");
}
__device__ __forceinline__ void wait_async0() {
    asm volatile("s_wait_asynccnt 0x0" ::: "memory");
}

#define GM   4096
#define GN   4096
#define GK   2048
#define KIH  1024
#define BM   128
#define BN   128
#define BK   32
#define LDSS 40   // padded LDS row stride in bf16 elems (80B, 16B-aligned)

// ---------------------------------------------------------------------------
// Kernel 0: f32 -> bf16 pre-conversion with K-concatenation.
//   blockIdx.y==0: Abf[4096][2048] = [x | hx]
//   blockIdx.y==1: Wbf[4096][2048] = [W_ih | W_hh]
// Bandwidth-bound one-shot pass (96 MB traffic ~ 4 us at 23.3 TB/s).
// ---------------------------------------------------------------------------
__global__ __launch_bounds__(256)
void convert_bf16_kernel(const float* __restrict__ x,  const float* __restrict__ hx,
                         const float* __restrict__ wih, const float* __restrict__ whh,
                         unsigned short* __restrict__ Abf,
                         unsigned short* __restrict__ Wbf)
{
    const int t   = blockIdx.x * 256 + threadIdx.x;  // 0 .. 4096*256-1
    const int row = t >> 8;
    const int c   = t & 255;          // 8-elem chunk within the 2048-wide row
    const int k8  = c * 8;

    const float* s0;
    const float* s1;
    unsigned short* dst;
    if (blockIdx.y == 0) { s0 = x;   s1 = hx;  dst = Abf; }
    else                 { s0 = wih; s1 = whh; dst = Wbf; }

    const float* src = (k8 < KIH) ? s0 : s1;
    const int    kk  = k8 & (KIH - 1);

    const float4 f0 = *(const float4*)(src + (size_t)row * KIH + kk);
    const float4 f1 = *(const float4*)(src + (size_t)row * KIH + kk + 4);

    uint4 o;
    o.x = f2bf(f0.x) | (f2bf(f0.y) << 16);
    o.y = f2bf(f0.z) | (f2bf(f0.w) << 16);
    o.z = f2bf(f1.x) | (f2bf(f1.y) << 16);
    o.w = f2bf(f1.z) | (f2bf(f1.w) << 16);
    *(uint4*)(dst + (size_t)row * GK + k8) = o;
}

// ---------------------------------------------------------------------------
// Kernel 1a (preferred): bf16 GEMM with async global->LDS staging.
//   gates[4096,4096] = Abf (4096x2048) @ Wbf^T (row-major [N,K])
//   BM=BN=128, BK=32, 8 waves (4x2), each wave: 2x4 fragments of 16x16.
//   K-loop does NO conversion VALU: staging is global_load_async_to_lds_b128,
//   consumption is ds_load_b128 fragment reads feeding v_wmma.
// ---------------------------------------------------------------------------
__global__ __launch_bounds__(256)
void gates_gemm_bf16_kernel(const unsigned short* __restrict__ Abf,
                            const unsigned short* __restrict__ Wbf,
                            float* __restrict__ gates)
{
    __shared__ __attribute__((aligned(16))) unsigned short lA[BM * LDSS];
    __shared__ __attribute__((aligned(16))) unsigned short lB[BN * LDSS];

    const int tid   = threadIdx.x;
    const int bm    = blockIdx.y;
    const int bn    = blockIdx.x;
    const int wave  = tid >> 5;
    const int lane  = tid & 31;
    const int laneL = lane & 15;
    const int laneH = lane >> 4;
    const int wm    = wave & 3;      // 4 wave rows  -> 128 M
    const int wn    = wave >> 2;     // 2 wave cols  -> 128 N

    const int rowA0 = bm * BM;
    const int rowB0 = bn * BN;

    // LDS byte offsets of the staging buffers (addrspace(3) offset fits 32b)
    const unsigned ldsA0 = (unsigned)(uintptr_t)(&lA[0]);
    const unsigned ldsB0 = (unsigned)(uintptr_t)(&lB[0]);

    // staging coords: 512 x 16B chunks per tile, 2 per thread
    const int r0 = tid >> 2,        c0 = tid & 3;
    const int r1 = (tid + 256) >> 2, c1 = (tid + 256) & 3;

    v8f acc[2][4];
    #pragma unroll
    for (int mi = 0; mi < 2; ++mi)
        #pragma unroll
        for (int ni = 0; ni < 4; ++ni)
            #pragma unroll
            for (int r = 0; r < 8; ++r)
                acc[mi][ni][r] = 0.0f;

    for (int k0 = 0; k0 < GK; k0 += BK) {
        __syncthreads();   // prev stage's LDS fully consumed (wmma waited dscnt)

        const unsigned kb = (unsigned)k0 * 2;   // byte offset along K
        // A tile: rows rowA0..+127, 64B per row
        async_ld16(ldsA0 + (unsigned)(r0 * LDSS * 2 + c0 * 16),
                   (unsigned)((rowA0 + r0) * (GK * 2)) + kb + c0 * 16, Abf);
        async_ld16(ldsA0 + (unsigned)(r1 * LDSS * 2 + c1 * 16),
                   (unsigned)((rowA0 + r1) * (GK * 2)) + kb + c1 * 16, Abf);
        // W tile
        async_ld16(ldsB0 + (unsigned)(r0 * LDSS * 2 + c0 * 16),
                   (unsigned)((rowB0 + r0) * (GK * 2)) + kb + c0 * 16, Wbf);
        async_ld16(ldsB0 + (unsigned)(r1 * LDSS * 2 + c1 * 16),
                   (unsigned)((rowB0 + r1) * (GK * 2)) + kb + c1 * 16, Wbf);

        wait_async0();
        __syncthreads();   // tile visible to all waves

        // prefetch next K tile into L2 (global_prefetch_b8)
        if (k0 + BK < GK) {
            __builtin_prefetch(Abf + (size_t)(rowA0 + r0) * GK + (k0 + BK) + c0 * 8, 0, 1);
            __builtin_prefetch(Wbf + (size_t)(rowB0 + r0) * GK + (k0 + BK) + c0 * 8, 0, 1);
        }

        // A (16x32): lanes 0-15 row M=laneL hold K 0..7 & 16..23;
        //            lanes 16-31 same rows hold K 8..15 & 24..31.
        FragAB a[2];
        #pragma unroll
        for (int mi = 0; mi < 2; ++mi) {
            const unsigned short* p =
                &lA[(wm * 32 + mi * 16 + laneL) * LDSS + laneH * 8];
            a[mi].q[0] = *(const uint4*)(p);
            a[mi].q[1] = *(const uint4*)(p + 16);
        }
        // B (32x16): lane n (0-15) holds column N=n (== W row n), K 0..7 & 16..23;
        //            lanes 16-31 hold same columns, K 8..15 & 24..31.
        FragAB b[4];
        #pragma unroll
        for (int ni = 0; ni < 4; ++ni) {
            const unsigned short* p =
                &lB[(wn * 64 + ni * 16 + laneL) * LDSS + laneH * 8];
            b[ni].q[0] = *(const uint4*)(p);
            b[ni].q[1] = *(const uint4*)(p + 16);
        }

        #pragma unroll
        for (int mi = 0; mi < 2; ++mi)
            #pragma unroll
            for (int ni = 0; ni < 4; ++ni)
                acc[mi][ni] = __builtin_amdgcn_wmma_f32_16x16x32_bf16(
                    false, a[mi].v, false, b[ni].v,
                    (short)0, acc[mi][ni], false, false);
    }

    #pragma unroll
    for (int mi = 0; mi < 2; ++mi)
        #pragma unroll
        for (int ni = 0; ni < 4; ++ni)
            #pragma unroll
            for (int r = 0; r < 8; ++r) {
                const int gm = rowA0 + wm * 32 + mi * 16 + laneH * 8 + r;
                const int gn = rowB0 + wn * 64 + ni * 16 + laneL;
                gates[(size_t)gm * GN + gn] = acc[mi][ni][r];
            }
}

// ---------------------------------------------------------------------------
// Kernel 1b (fallback if ws too small for bf16 staging buffers):
//   previous round's kernel with in-loop f32->bf16 conversion.
// ---------------------------------------------------------------------------
__global__ __launch_bounds__(256)
void gates_gemm_f32src_kernel(const float* __restrict__ x,
                              const float* __restrict__ hx,
                              const float* __restrict__ wih,
                              const float* __restrict__ whh,
                              float* __restrict__ gates)
{
    __shared__ __attribute__((aligned(16))) unsigned short lA[BM * LDSS];
    __shared__ __attribute__((aligned(16))) unsigned short lB[BN * LDSS];

    const int tid   = threadIdx.x;
    const int bm    = blockIdx.y;
    const int bn    = blockIdx.x;
    const int wave  = tid >> 5;
    const int lane  = tid & 31;
    const int laneL = lane & 15;
    const int laneH = lane >> 4;
    const int wm    = wave & 3;
    const int wn    = wave >> 2;

    const int rowA0 = bm * BM;
    const int rowB0 = bn * BN;

    v8f acc[2][4];
    #pragma unroll
    for (int mi = 0; mi < 2; ++mi)
        #pragma unroll
        for (int ni = 0; ni < 4; ++ni)
            #pragma unroll
            for (int r = 0; r < 8; ++r)
                acc[mi][ni][r] = 0.0f;

    for (int k0 = 0; k0 < GK; k0 += BK) {
        const float* Asrc = (k0 < KIH) ? x   : hx;
        const float* Wsrc = (k0 < KIH) ? wih : whh;
        const int    kc   = (k0 < KIH) ? k0  : (k0 - KIH);

        float4 ra[4], rw[4];
        #pragma unroll
        for (int i = 0; i < 4; ++i) {
            const int idx = tid + i * 256;
            const int r   = idx >> 3;
            const int c4  = idx & 7;
            ra[i] = *(const float4*)(Asrc + (size_t)(rowA0 + r) * KIH + kc + c4 * 4);
            rw[i] = *(const float4*)(Wsrc + (size_t)(rowB0 + r) * KIH + kc + c4 * 4);
        }

        __syncthreads();
        #pragma unroll
        for (int i = 0; i < 4; ++i) {
            const int idx = tid + i * 256;
            const int r   = idx >> 3;
            const int c4  = idx & 7;
            uint2 pa, pw;
            pa.x = f2bf(ra[i].x) | (f2bf(ra[i].y) << 16);
            pa.y = f2bf(ra[i].z) | (f2bf(ra[i].w) << 16);
            pw.x = f2bf(rw[i].x) | (f2bf(rw[i].y) << 16);
            pw.y = f2bf(rw[i].z) | (f2bf(rw[i].w) << 16);
            *(uint2*)&lA[r * LDSS + c4 * 4] = pa;
            *(uint2*)&lB[r * LDSS + c4 * 4] = pw;
        }
        __syncthreads();

        FragAB a[2];
        #pragma unroll
        for (int mi = 0; mi < 2; ++mi) {
            const unsigned short* p =
                &lA[(wm * 32 + mi * 16 + laneL) * LDSS + laneH * 8];
            a[mi].q[0] = *(const uint4*)(p);
            a[mi].q[1] = *(const uint4*)(p + 16);
        }
        FragAB b[4];
        #pragma unroll
        for (int ni = 0; ni < 4; ++ni) {
            const unsigned short* p =
                &lB[(wn * 64 + ni * 16 + laneL) * LDSS + laneH * 8];
            b[ni].q[0] = *(const uint4*)(p);
            b[ni].q[1] = *(const uint4*)(p + 16);
        }

        #pragma unroll
        for (int mi = 0; mi < 2; ++mi)
            #pragma unroll
            for (int ni = 0; ni < 4; ++ni)
                acc[mi][ni] = __builtin_amdgcn_wmma_f32_16x16x32_bf16(
                    false, a[mi].v, false, b[ni].v,
                    (short)0, acc[mi][ni], false, false);
    }

    #pragma unroll
    for (int mi = 0; mi < 2; ++mi)
        #pragma unroll
        for (int ni = 0; ni < 4; ++ni)
            #pragma unroll
            for (int r = 0; r < 8; ++r) {
                const int gm = rowA0 + wm * 32 + mi * 16 + laneH * 8 + r;
                const int gn = rowB0 + wn * 64 + ni * 16 + laneL;
                gates[(size_t)gm * GN + gn] = acc[mi][ni][r];
            }
}

// ---------------------------------------------------------------------------
// Kernel 2: fused LayerNorm-LSTM epilogue, one block per batch row
// ---------------------------------------------------------------------------
#define HB 1024
#define OUT_PLANE (4096 * 1024)

__device__ __forceinline__ void block_reduce2(float& s0, float& s1, float* red)
{
    #pragma unroll
    for (int off = 16; off > 0; off >>= 1) {
        s0 += __shfl_xor(s0, off, 32);
        s1 += __shfl_xor(s1, off, 32);
    }
    const int wid = threadIdx.x >> 5;
    if ((threadIdx.x & 31) == 0) { red[wid] = s0; red[8 + wid] = s1; }
    __syncthreads();
    if (threadIdx.x == 0) {
        float a = 0.f, c = 0.f;
        #pragma unroll
        for (int i = 0; i < 8; ++i) { a += red[i]; c += red[8 + i]; }
        red[16] = a; red[17] = c;
    }
    __syncthreads();
    s0 = red[16];
    s1 = red[17];
    __syncthreads();
}

__device__ __forceinline__ void layernorm4(float xv[4],
                                           const float* __restrict__ gm,
                                           const float* __restrict__ bt,
                                           const int h[4], float* red)
{
    float s0 = 0.f, s1 = 0.f;
    #pragma unroll
    for (int j = 0; j < 4; ++j) { s0 += xv[j]; s1 += xv[j] * xv[j]; }
    block_reduce2(s0, s1, red);
    const float mean = s0 * (1.0f / (float)HB);
    float var = s1 * (1.0f / (float)HB) - mean * mean;
    const float filt = (var < 1e-12f) ? 0.01f : 0.0f;
    const float rstd = rsqrtf(var + 1e-12f + filt);
    #pragma unroll
    for (int j = 0; j < 4; ++j)
        xv[j] = (xv[j] - mean) * rstd * gm[h[j]] + bt[h[j]];
}

__device__ __forceinline__ float sigmoidf_(float x)
{
    return 1.0f / (1.0f + __expf(-x));
}

__global__ __launch_bounds__(256)
void lstm_epilogue_kernel(const float* __restrict__ gates,
                          const float* __restrict__ cx,
                          const float* __restrict__ gamma,
                          const float* __restrict__ beta,
                          float* __restrict__ out)
{
    __shared__ float red[18];
    const int b   = blockIdx.x;
    const int tid = threadIdx.x;

    int h[4];
    #pragma unroll
    for (int j = 0; j < 4; ++j) h[j] = tid + j * 256;

    const float* g = gates + (size_t)b * (4 * HB);

    float v0[4], v1[4], v2[4], v3[4];
    #pragma unroll
    for (int j = 0; j < 4; ++j) {
        v0[j] = g[0 * HB + h[j]];
        v1[j] = g[1 * HB + h[j]];
        v2[j] = g[2 * HB + h[j]];
        v3[j] = g[3 * HB + h[j]];
    }

    layernorm4(v0, gamma + 0 * HB, beta + 0 * HB, h, red);
    layernorm4(v1, gamma + 1 * HB, beta + 1 * HB, h, red);
    layernorm4(v2, gamma + 2 * HB, beta + 2 * HB, h, red);
    layernorm4(v3, gamma + 3 * HB, beta + 3 * HB, h, red);

    float cyv[4], og[4];
    #pragma unroll
    for (int j = 0; j < 4; ++j) {
        const float ing = sigmoidf_(v0[j]);
        const float fg  = sigmoidf_(v1[j] + 1.0f);   // forget bias
        const float cg  = tanhf(v2[j]);
        og[j] = sigmoidf_(v3[j]);
        cyv[j] = fg * cx[(size_t)b * HB + h[j]] + ing * cg;
    }

    layernorm4(cyv, gamma + 4 * HB, beta + 4 * HB, h, red);

    #pragma unroll
    for (int j = 0; j < 4; ++j) {
        const float cy = cyv[j];
        const float hy = og[j] * tanhf(cy);
        const size_t o = (size_t)b * HB + h[j];
        out[o] = hy;                    // hy_dropped (dropout p=0)
        out[OUT_PLANE + o] = hy;        // hy
        out[2 * (size_t)OUT_PLANE + o] = cy;
    }
}

// ---------------------------------------------------------------------------
// Launcher
// ---------------------------------------------------------------------------
extern "C" void kernel_launch(void* const* d_in, const int* in_sizes, int n_in,
                              void* d_out, int out_size, void* d_ws, size_t ws_size,
                              hipStream_t stream)
{
    const float* x     = (const float*)d_in[0];
    const float* hx    = (const float*)d_in[1];
    const float* cx    = (const float*)d_in[2];
    const float* wih   = (const float*)d_in[3];
    const float* whh   = (const float*)d_in[4];
    const float* gamma = (const float*)d_in[5];
    const float* beta  = (const float*)d_in[6];
    float* out = (float*)d_out;

    const size_t bf16Bytes  = (size_t)GM * GK * sizeof(unsigned short); // 16 MB each
    const size_t gatesBytes = (size_t)GM * GN * sizeof(float);          // 64 MB
    const dim3 ggrid(GN / BN, GM / BM);                                 // 32 x 32

    if (ws_size >= 2 * bf16Bytes + gatesBytes) {
        // Preferred path: pre-converted bf16 operands + async-LDS GEMM.
        unsigned short* Abf = (unsigned short*)d_ws;
        unsigned short* Wbf = (unsigned short*)((char*)d_ws + bf16Bytes);
        float* gates        = (float*)((char*)d_ws + 2 * bf16Bytes);

        convert_bf16_kernel<<<dim3(4096, 2), 256, 0, stream>>>(x, hx, wih, whh, Abf, Wbf);
        gates_gemm_bf16_kernel<<<ggrid, 256, 0, stream>>>(Abf, Wbf, gates);
        lstm_epilogue_kernel<<<4096, 256, 0, stream>>>(gates, cx, gamma, beta, out);
    } else {
        // Fallback: convert inside the GEMM loop, gates only in ws.
        float* gates = (float*)d_ws;
        gates_gemm_f32src_kernel<<<ggrid, 256, 0, stream>>>(x, hx, wih, whh, gates);
        lstm_epilogue_kernel<<<4096, 256, 0, stream>>>(gates, cx, gamma, beta, out);
    }
}
